// CrossPredictor_11055245820132
// MI455X (gfx1250) — compile-verified, matrix operand order checked by reading, not run on hardware
//
#include <hip/hip_runtime.h>
#include <math.h>

#define CC 1024
#define HH 16
#define DHH 64

typedef __attribute__((ext_vector_type(16))) __bf16 v16bf;
typedef __attribute__((ext_vector_type(8)))  __bf16 v8bf;
typedef __attribute__((ext_vector_type(8)))  float  v8f;

// ---------------------------------------------------------------------------
// CDNA5 async global->LDS copy (ASYNCcnt path).  A generic pointer into LDS
// carries the LDS byte offset in its low 32 bits (ISA 10.2: LDS aperture
// address = addr[31:0]), so we can hand it straight to the instruction.
// ---------------------------------------------------------------------------
static __device__ __forceinline__ void async_copy16(void* lds_dst, const void* gsrc) {
  unsigned ldsa = (unsigned)(uintptr_t)lds_dst;
  unsigned long long ga = (unsigned long long)(uintptr_t)gsrc;
  asm volatile("global_load_async_to_lds_b128 %0, %1, off"
               :: "v"(ldsa), "v"(ga)
               : "memory");
}
static __device__ __forceinline__ void async_wait0() {
  asm volatile("s_wait_asynccnt 0x0" ::: "memory");
}

// ---------------------------------------------------------------------------
// Fragment helpers (CDNA5 wave32 WMMA 16x16x32 bf16)
// A-fragment (16xK row-major source): lanes 0-15 hold row r=lane, elems 0..7 =
// K 0..7, elems 8..15 = K 16..23; lanes 16-31 hold the K 8..15 / 24..31 halves.
// The B-fragment is the same pattern applied to B^T, so we store all "B"
// operands transposed ([N][K] row-major) and reuse one loader.
// ---------------------------------------------------------------------------
static __device__ __forceinline__ v16bf lds_frag(const __bf16* base, int ld, int lane) {
  const int r  = lane & 15;
  const int kh = (lane >> 4) << 3;            // 0 or 8
  const __bf16* p = base + r * ld + kh;
  v8bf lo = *(const v8bf*)(p);                // 16B ds_load
  v8bf hi = *(const v8bf*)(p + 16);
  v16bf f;
#pragma unroll
  for (int i = 0; i < 8; ++i) { f[i] = lo[i]; f[i + 8] = hi[i]; }
  return f;
}

static __device__ __forceinline__ v8f wmma_bf16(v16bf a, v16bf b, v8f c) {
  return __builtin_amdgcn_wmma_f32_16x16x32_bf16(false, a, false, b, (short)0, c,
                                                 false, false);
}

static __device__ __forceinline__ v8f vzero8() {
  v8f z;
#pragma unroll
  for (int e = 0; e < 8; ++e) z[e] = 0.f;
  return z;
}

// ---------------------------------------------------------------------------
// Block reduction over 256 threads (8 waves of 32)
// ---------------------------------------------------------------------------
static __device__ __forceinline__ float block_sum256(float v, float* red) {
  const int tid = threadIdx.x;
#pragma unroll
  for (int off = 16; off > 0; off >>= 1) v += __shfl_xor(v, off, 32);
  if ((tid & 31) == 0) red[tid >> 5] = v;
  __syncthreads();
  if (tid == 0) {
    float s = 0.f;
#pragma unroll
    for (int i = 0; i < 8; ++i) s += red[i];
    red[0] = s;
  }
  __syncthreads();
  float r = red[0];
  __syncthreads();
  return r;
}

// ---------------------------------------------------------------------------
// Kernel 1: pos-enc (computed on the fly) + LayerNorm over C, channels-first
// input [B, C, T] f32 -> row-major [B*T, C] bf16 (and optional f32 copy)
// ---------------------------------------------------------------------------
__global__ __launch_bounds__(256)
void prep_posln(const float* __restrict__ x, const float* __restrict__ g,
                const float* __restrict__ bta, __bf16* __restrict__ outB,
                float* __restrict__ outF, int T) {
  __shared__ float red[8];
  const int bt = blockIdx.x;
  const int bb = bt / T;
  const int t  = bt - bb * T;
  const int tid = threadIdx.x;
  float v[4];
  float sum = 0.f;
#pragma unroll
  for (int k = 0; k < 4; ++k) {
    const int c   = tid + (k << 8);
    const float f = __expf(-(float)(c & ~1) * (9.210340372f / 1024.f));
    const float a = (float)t * f;
    const float pe = (c & 1) ? __cosf(a) : __sinf(a);
    v[k] = x[((size_t)bb * CC + c) * T + t] + pe;
    sum += v[k];
  }
  const float mu = block_sum256(sum, red) * (1.f / 1024.f);
  float vs = 0.f;
#pragma unroll
  for (int k = 0; k < 4; ++k) { const float d = v[k] - mu; vs += d * d; }
  const float var = block_sum256(vs, red) * (1.f / 1024.f);
  const float inv = rsqrtf(var + 1e-5f);
#pragma unroll
  for (int k = 0; k < 4; ++k) {
    const int c = tid + (k << 8);
    const float y = (v[k] - mu) * inv * g[c] + bta[c];
    outB[(size_t)bt * CC + c] = (__bf16)y;
    if (outF) outF[(size_t)bt * CC + c] = y;
  }
}

// ---------------------------------------------------------------------------
// Kernel 2: LayerNorm of a row-major f32 [M, C] matrix -> bf16 (FFN pre-LN)
// ---------------------------------------------------------------------------
__global__ __launch_bounds__(256)
void row_ln(const float* __restrict__ x, const float* __restrict__ g,
            const float* __restrict__ bta, __bf16* __restrict__ outB) {
  __shared__ float red[8];
  const int row = blockIdx.x;
  const int tid = threadIdx.x;
  float v[4];
  float sum = 0.f;
#pragma unroll
  for (int k = 0; k < 4; ++k) {
    const int c = tid + (k << 8);
    v[k] = x[(size_t)row * CC + c];
    sum += v[k];
  }
  const float mu = block_sum256(sum, red) * (1.f / 1024.f);
  float vs = 0.f;
#pragma unroll
  for (int k = 0; k < 4; ++k) { const float d = v[k] - mu; vs += d * d; }
  const float var = block_sum256(vs, red) * (1.f / 1024.f);
  const float inv = rsqrtf(var + 1e-5f);
#pragma unroll
  for (int k = 0; k < 4; ++k) {
    const int c = tid + (k << 8);
    outB[(size_t)row * CC + c] = (__bf16)((v[k] - mu) * inv * g[c] + bta[c]);
  }
}

// ---------------------------------------------------------------------------
// Kernel 3: weight convert f32 [K, N] -> bf16 transposed [N, K]
// ---------------------------------------------------------------------------
__global__ __launch_bounds__(256)
void wtrans(const float* __restrict__ W, __bf16* __restrict__ Wt, int K, int N) {
  __shared__ float tile[32][33];
  const int n0 = blockIdx.x * 32;
  const int k0 = blockIdx.y * 32;
  const int tx = threadIdx.x & 31;
  const int ty = threadIdx.x >> 5;
#pragma unroll
  for (int i = ty; i < 32; i += 8)
    tile[i][tx] = W[(size_t)(k0 + i) * N + n0 + tx];
  __syncthreads();
#pragma unroll
  for (int i = ty; i < 32; i += 8)
    Wt[(size_t)(n0 + i) * K + k0 + tx] = (__bf16)tile[tx][i];
}

// ---------------------------------------------------------------------------
// Kernel 4: LDS-staged bf16 WMMA GEMM.  C[M,N] = A[M,K] @ Bt[N,K]^T
// Block tile 128x256, 8 waves in a 2x4 grid, each wave 64x64 (16 accums).
// Tiles are staged with GLOBAL_LOAD_ASYNC_TO_LDS_B128 (ASYNCcnt path).
// Epilogues:
//   0: store bf16                      (QKV projections)
//   1: + residual, store f32           (Wo: r = y + q)
//   2: + bias, exact GELU, store bf16  (FFN1)
//   3: + bias + residual, store f32    (FFN2 + r)
// Assumes M%128==0, N%256==0, K%32==0 (true for all uses here).
// ---------------------------------------------------------------------------
template <int EPI>
__global__ __launch_bounds__(256)
void gemm_bf16(const __bf16* __restrict__ A, const __bf16* __restrict__ Bt,
               const float* __restrict__ bias, const float* __restrict__ res,
               float* __restrict__ outF, __bf16* __restrict__ outB,
               int M, int N, int K) {
  constexpr int LDT = 40;                      // halfs; 80B rows keep 16B align
  __shared__ __bf16 As[128 * LDT];
  __shared__ __bf16 Bs[256 * LDT];
  const int tid  = threadIdx.x;
  const int wave = tid >> 5;
  const int lane = tid & 31;
  const int m0 = blockIdx.y * 128;
  const int n0 = blockIdx.x * 256;
  const int wm = wave >> 2;                    // 0..1 (64 rows each)
  const int wn = wave & 3;                     // 0..3 (64 cols each)

  v8f acc[4][4];
#pragma unroll
  for (int i = 0; i < 4; ++i)
#pragma unroll
    for (int j = 0; j < 4; ++j) acc[i][j] = vzero8();

  for (int k0 = 0; k0 < K; k0 += 32) {
    // async-stage 128x32 of A (512 16B chunks) and 256x32 of Bt (1024 chunks)
#pragma unroll
    for (int c = tid; c < 512; c += 256) {
      const int row = c >> 2, col = (c & 3) << 3;
      async_copy16(&As[row * LDT + col],
                   &A[(size_t)(m0 + row) * K + k0 + col]);
      if (k0 + 32 < K)                          // hint next K-slice into L2
        __builtin_prefetch(&A[(size_t)(m0 + row) * K + k0 + 32 + col], 0, 1);
    }
#pragma unroll
    for (int c = tid; c < 1024; c += 256) {
      const int row = c >> 2, col = (c & 3) << 3;
      async_copy16(&Bs[row * LDT + col],
                   &Bt[(size_t)(n0 + row) * K + k0 + col]);
    }
    async_wait0();
    __syncthreads();
    v16bf bfrag[4];
#pragma unroll
    for (int j = 0; j < 4; ++j)
      bfrag[j] = lds_frag(&Bs[(wn * 64 + j * 16) * LDT], LDT, lane);
#pragma unroll
    for (int i = 0; i < 4; ++i) {
      const v16bf afrag = lds_frag(&As[(wm * 64 + i * 16) * LDT], LDT, lane);
#pragma unroll
      for (int j = 0; j < 4; ++j) acc[i][j] = wmma_bf16(afrag, bfrag[j], acc[i][j]);
    }
    __syncthreads();
  }

  // epilogue: acc elem e -> (row = base + e, col = ln) per C/D VGPR layout
  const int ln = lane & 15;
  const int lm = (lane >> 4) << 3;
#pragma unroll
  for (int i = 0; i < 4; ++i) {
#pragma unroll
    for (int j = 0; j < 4; ++j) {
      const int gmb = m0 + wm * 64 + i * 16 + lm;
      const int gn  = n0 + wn * 64 + j * 16 + ln;
      float bv = 0.f;
      if (EPI == 2 || EPI == 3) bv = bias[gn];
#pragma unroll
      for (int e = 0; e < 8; ++e) {
        const int gm = gmb + e;
        float v = acc[i][j][e];
        if (EPI == 2 || EPI == 3) v += bv;
        if (EPI == 2) v = 0.5f * v * (1.f + erff(v * 0.70710678f));
        if (EPI == 1 || EPI == 3) v += res[(size_t)gm * N + gn];
        if (EPI == 0 || EPI == 2) outB[(size_t)gm * N + gn] = (__bf16)v;
        else                      outF[(size_t)gm * N + gn] = v;
      }
    }
  }
}

// ---------------------------------------------------------------------------
// Kernel 5: flash attention.  Per block: one (b, h) and a 64-query tile.
// Q and K tiles are async-staged; V is staged transposed so V^T reads as an
// A-pattern fragment.  The 1/sqrt(DH) scale is applied to S accumulators.
// ---------------------------------------------------------------------------
__global__ __launch_bounds__(256)
void flash_attn(const __bf16* __restrict__ Qg, const __bf16* __restrict__ Kg,
                const __bf16* __restrict__ Vg, __bf16* __restrict__ ctxO,
                int Tq, int Tk) {
  constexpr int LDB  = 72;                     // bf16 ld: 144B rows, 16B align
  constexpr int LDSF = 68;                     // f32 S ld
  __shared__ __bf16 Qs [64 * LDB];
  __shared__ __bf16 Ks [64 * LDB];
  __shared__ __bf16 Vts[64 * LDB];
  __shared__ __bf16 Ps [64 * LDB];
  __shared__ float  Ss [64 * LDSF];
  __shared__ float  mS[64], lS[64], aS[64];

  const int bh   = blockIdx.x;
  const int b    = bh >> 4;                    // H = 16
  const int h    = bh & 15;
  const int q0   = blockIdx.y * 64;
  const int tid  = threadIdx.x;
  const int wave = tid >> 5;
  const int lane = tid & 31;
  const int colh = h * DHH;
  const size_t qrow0 = (size_t)b * Tq + q0;
  const size_t krow0 = (size_t)b * Tk;

  // async-stage Q tile (64x64)
  for (int c = tid; c < 512; c += 256) {
    const int r = c >> 3, cs = (c & 7) << 3;
    async_copy16(&Qs[r * LDB + cs], &Qg[(qrow0 + r) * CC + colh + cs]);
  }
  if (tid < 64) { mS[tid] = -1e30f; lS[tid] = 0.f; }

  v8f acc[2];
  acc[0] = vzero8();
  acc[1] = vzero8();
  __syncthreads();

  for (int k0 = 0; k0 < Tk; k0 += 64) {
    // stage K (async, row-major) and V (transposed scatter)
    for (int c = tid; c < 512; c += 256) {
      const int r = c >> 3, cs = (c & 7) << 3;
      const size_t src = (krow0 + k0 + r) * CC + colh + cs;
      async_copy16(&Ks[r * LDB + cs], &Kg[src]);
      v8bf vv = *(const v8bf*)&Vg[src];
#pragma unroll
      for (int e = 0; e < 8; ++e) Vts[(cs + e) * LDB + r] = vv[e];
    }
    async_wait0();
    __syncthreads();

    // S tile 64x64: 16 fragment tiles, 2 per wave; scale by 1/sqrt(DH)
#pragma unroll
    for (int s = 0; s < 2; ++s) {
      const int flat = wave * 2 + s, mi = flat >> 2, ni = flat & 3;
      v8f sa = vzero8();
#pragma unroll
      for (int d = 0; d < 64; d += 32) {
        const v16bf af = lds_frag(&Qs[(mi * 16) * LDB + d], LDB, lane);
        const v16bf bf = lds_frag(&Ks[(ni * 16) * LDB + d], LDB, lane);
        sa = wmma_bf16(af, bf, sa);
      }
      const int rW = mi * 16 + ((lane >> 4) << 3);
      const int cW = ni * 16 + (lane & 15);
#pragma unroll
      for (int e = 0; e < 8; ++e) Ss[(rW + e) * LDSF + cW] = sa[e] * 0.125f;
    }
    __syncthreads();

    // online softmax, one thread per query row
    if (tid < 64) {
      const float mOld = mS[tid];
      float mt = mOld;
      for (int c = 0; c < 64; ++c) mt = fmaxf(mt, Ss[tid * LDSF + c]);
      const float alpha = __expf(mOld - mt);
      float lsum = 0.f;
      for (int c = 0; c < 64; ++c) {
        const float p = __expf(Ss[tid * LDSF + c] - mt);
        lsum += p;
        Ps[tid * LDB + c] = (__bf16)p;
      }
      mS[tid] = mt;
      lS[tid] = lS[tid] * alpha + lsum;
      aS[tid] = alpha;
    }
    __syncthreads();

    // rescale running ctx, then ctx += P @ V   (B^T = V^T = Vts rows)
#pragma unroll
    for (int s = 0; s < 2; ++s) {
      const int flat = wave * 2 + s, mi = flat >> 2, ni = flat & 3;
      const int rB = mi * 16 + ((lane >> 4) << 3);
#pragma unroll
      for (int e = 0; e < 8; ++e) acc[s][e] *= aS[rB + e];
#pragma unroll
      for (int kk = 0; kk < 64; kk += 32) {
        const v16bf af = lds_frag(&Ps[(mi * 16) * LDB + kk], LDB, lane);
        const v16bf bf = lds_frag(&Vts[(ni * 16) * LDB + kk], LDB, lane);
        acc[s] = wmma_bf16(af, bf, acc[s]);
      }
    }
    __syncthreads();
  }

  // finalize: divide by softmax denominator, store ctx head slice
#pragma unroll
  for (int s = 0; s < 2; ++s) {
    const int flat = wave * 2 + s, mi = flat >> 2, ni = flat & 3;
    const int rB = mi * 16 + ((lane >> 4) << 3);
    const int cc = ni * 16 + (lane & 15);
#pragma unroll
    for (int e = 0; e < 8; ++e) {
      const int r = rB + e;
      ctxO[(qrow0 + r) * CC + colh + cc] = (__bf16)(acc[s][e] / lS[r]);
    }
  }
}

// ---------------------------------------------------------------------------
// Kernel 6: final transpose f32 [B*T, C] -> [B, C, T]
// ---------------------------------------------------------------------------
__global__ __launch_bounds__(256)
void out_trans(const float* __restrict__ in, float* __restrict__ out, int T) {
  __shared__ float tile[32][33];
  const int b  = blockIdx.z;
  const int t0 = blockIdx.x * 32;
  const int c0 = blockIdx.y * 32;
  const int tx = threadIdx.x & 31;
  const int ty = threadIdx.x >> 5;
#pragma unroll
  for (int i = ty; i < 32; i += 8)
    tile[i][tx] = in[((size_t)b * T + t0 + i) * CC + c0 + tx];
  __syncthreads();
#pragma unroll
  for (int i = ty; i < 32; i += 8)
    out[((size_t)b * CC + c0 + i) * T + t0 + tx] = tile[tx][i];
}

// ---------------------------------------------------------------------------
// Host orchestration
// ---------------------------------------------------------------------------
extern "C" void kernel_launch(void* const* d_in, const int* in_sizes, int n_in,
                              void* d_out, int out_size, void* d_ws, size_t ws_size,
                              hipStream_t stream) {
  (void)in_sizes; (void)n_in; (void)out_size; (void)ws_size;
  const int B = 2, T = 2048, M = B * T;

  const float* zt   = (const float*)d_in[0];
  const float* za   = (const float*)d_in[1];
  const float* Wq   = (const float*)d_in[2];
  const float* Wk   = (const float*)d_in[3];
  const float* Wv   = (const float*)d_in[4];
  const float* Wo   = (const float*)d_in[5];
  const float* lnqg = (const float*)d_in[6];
  const float* lnqb = (const float*)d_in[7];
  const float* lnkg = (const float*)d_in[8];
  const float* lnkb = (const float*)d_in[9];
  const float* flng = (const float*)d_in[10];
  const float* flnb = (const float*)d_in[11];
  const float* W1   = (const float*)d_in[12];
  const float* b1   = (const float*)d_in[13];
  const float* W2   = (const float*)d_in[14];
  const float* b2   = (const float*)d_in[15];
  float* out = (float*)d_out;

  char* ws = (char*)d_ws;
  size_t off = 0;
  auto alloc = [&](size_t bytes) -> void* {
    void* p = ws + off;
    off += (bytes + 255) & ~(size_t)255;
    return p;
  };
  __bf16* lnq_b  = (__bf16*)alloc((size_t)M * CC * 2);
  __bf16* lnkv_b = (__bf16*)alloc((size_t)M * CC * 2);
  float*  lnq_f  = (float*) alloc((size_t)M * CC * 4);
  __bf16* WqT    = (__bf16*)alloc((size_t)CC * CC * 2);
  __bf16* WkT    = (__bf16*)alloc((size_t)CC * CC * 2);
  __bf16* WvT    = (__bf16*)alloc((size_t)CC * CC * 2);
  __bf16* WoT    = (__bf16*)alloc((size_t)CC * CC * 2);
  __bf16* W1T    = (__bf16*)alloc((size_t)2 * CC * CC * 2);
  __bf16* W2T    = (__bf16*)alloc((size_t)2 * CC * CC * 2);
  __bf16* Qb     = (__bf16*)alloc((size_t)M * CC * 2);
  __bf16* Kb     = (__bf16*)alloc((size_t)M * CC * 2);
  __bf16* Vb     = (__bf16*)alloc((size_t)M * CC * 2);
  __bf16* ctxb   = (__bf16*)alloc((size_t)M * CC * 2);
  float*  r_f    = (float*) alloc((size_t)M * CC * 4);
  __bf16* h_b    = (__bf16*)alloc((size_t)M * CC * 2);
  __bf16* g1_b   = (__bf16*)alloc((size_t)M * 2 * CC * 2);
  float*  fin_f  = (float*) alloc((size_t)M * CC * 4);

  // 1) pos-enc + LN for q (bf16 + f32 residual copy) and kv (bf16 only)
  prep_posln<<<M, 256, 0, stream>>>(zt, lnqg, lnqb, lnq_b, lnq_f, T);
  prep_posln<<<M, 256, 0, stream>>>(za, lnkg, lnkb, lnkv_b, nullptr, T);

  // 2) weights -> bf16, transposed [N, K]
  wtrans<<<dim3(32, 32), 256, 0, stream>>>(Wq, WqT, CC, CC);
  wtrans<<<dim3(32, 32), 256, 0, stream>>>(Wk, WkT, CC, CC);
  wtrans<<<dim3(32, 32), 256, 0, stream>>>(Wv, WvT, CC, CC);
  wtrans<<<dim3(32, 32), 256, 0, stream>>>(Wo, WoT, CC, CC);
  wtrans<<<dim3(64, 32), 256, 0, stream>>>(W1, W1T, CC, 2 * CC);      // [1024,2048]->[2048,1024]
  wtrans<<<dim3(32, 64), 256, 0, stream>>>(W2, W2T, 2 * CC, CC);      // [2048,1024]->[1024,2048]

  // 3) QKV projections (bf16 out)
  const dim3 gemmB(256);
  gemm_bf16<0><<<dim3(CC / 256, M / 128), gemmB, 0, stream>>>(
      lnq_b,  WqT, nullptr, nullptr, nullptr, Qb, M, CC, CC);
  gemm_bf16<0><<<dim3(CC / 256, M / 128), gemmB, 0, stream>>>(
      lnkv_b, WkT, nullptr, nullptr, nullptr, Kb, M, CC, CC);
  gemm_bf16<0><<<dim3(CC / 256, M / 128), gemmB, 0, stream>>>(
      lnkv_b, WvT, nullptr, nullptr, nullptr, Vb, M, CC, CC);

  // 4) attention: grid (B*H, Tq/64)
  flash_attn<<<dim3(B * HH, T / 64), 256, 0, stream>>>(Qb, Kb, Vb, ctxb, T, T);

  // 5) output projection + residual: r = ctx @ Wo + lnq
  gemm_bf16<1><<<dim3(CC / 256, M / 128), gemmB, 0, stream>>>(
      ctxb, WoT, nullptr, lnq_f, r_f, nullptr, M, CC, CC);

  // 6) FFN pre-LN
  row_ln<<<M, 256, 0, stream>>>(r_f, flng, flnb, h_b);

  // 7) FFN1: gelu(h @ W1 + b1) -> bf16
  gemm_bf16<2><<<dim3(2 * CC / 256, M / 128), gemmB, 0, stream>>>(
      h_b, W1T, b1, nullptr, nullptr, g1_b, M, 2 * CC, CC);

  // 8) FFN2 + residual: fin = g1 @ W2 + b2 + r
  gemm_bf16<3><<<dim3(CC / 256, M / 128), gemmB, 0, stream>>>(
      g1_b, W2T, b2, r_f, fin_f, nullptr, M, CC, 2 * CC);

  // 9) back to channels-first [B, C, T]
  out_trans<<<dim3(T / 32, CC / 32, B), 256, 0, stream>>>(fin_f, out, T);
}
